// DTVNet_61220463837791
// MI455X (gfx1250) — compile-verified, matrix operand order checked by reading, not run on hardware
//
#include <hip/hip_runtime.h>
#include <stdint.h>

// ---------------------------------------------------------------------------
// DTVNet cascade for MI455X (gfx1250): bf16 WMMA implicit-GEMM for the two
// 32->32 3x3x3 convolutions, async global<->LDS staging, VALU for cheap ops.
// ---------------------------------------------------------------------------

#define V   96
#define VV  (V * V)
#define VOL (V * V * V)
#define C4  32
#define TIN 18  // 16 + 2 halo

typedef __bf16 bf16x16 __attribute__((ext_vector_type(16)));
typedef float  f32x8   __attribute__((ext_vector_type(8)));
typedef unsigned short u16;

// gfx1250 async global<->LDS (ASYNCcnt-tracked) if the toolchain exposes it
#if defined(__has_builtin)
#  if __has_builtin(__builtin_amdgcn_global_load_async_to_lds_b128) && \
      __has_builtin(__builtin_amdgcn_s_wait_asynccnt)
#    define USE_ASYNC_LD 1
#  endif
#  if __has_builtin(__builtin_amdgcn_global_store_async_from_lds_b128) && \
      __has_builtin(__builtin_amdgcn_s_wait_asynccnt)
#    define USE_ASYNC_ST 1
#  endif
#endif

#if defined(USE_ASYNC_LD) || defined(USE_ASYNC_ST)
// builtin prototype (from clang diagnostic): (int4 AS1*, int4 AS3*, Imm, Imm)
typedef int v4i __attribute__((ext_vector_type(4)));
typedef __attribute__((address_space(1))) v4i* g_v4i_p;  // global int4*
typedef __attribute__((address_space(3))) v4i* l_v4i_p;  // LDS int4*
#endif

__device__ __forceinline__ u16 f2bf(float f) {
    union { float f; unsigned u; } v; v.f = f;
    unsigned r = v.u + 0x7FFFu + ((v.u >> 16) & 1u);  // round-to-nearest-even
    return (u16)(r >> 16);
}
__device__ __forceinline__ float bf2f(u16 h) {
    union { unsigned u; float f; } v; v.u = ((unsigned)h) << 16;
    return v.f;
}

// ---------------------------------------------------------------------------
// t = image; p = q = s = 0
__global__ void k_init(const float* __restrict__ img, float* t, float* p, float* q, float* s) {
    int i = blockIdx.x * 256 + threadIdx.x;
    if (i < VOL) { t[i] = img[i]; p[i] = 0.f; q[i] = 0.f; s[i] = 0.f; }
}

// r[y][x] = sino[y][x] - sum_z t[z][y][x]
__global__ void k_residual(const float* __restrict__ t, const float* __restrict__ sino,
                           float* __restrict__ r) {
    int i = blockIdx.x * 256 + threadIdx.x;
    if (i >= VV) return;
    float acc = 0.f;
    for (int zz = 0; zz < V; ++zz) acc += t[zz * VV + i];
    r[i] = sino[i] - acc;
}

// z = t + lam[c] * r / V   (FDK broadcast over depth)
__global__ void k_z(const float* __restrict__ t, const float* __restrict__ r,
                    const float* __restrict__ lam, int c, float* __restrict__ z) {
    int i = blockIdx.x * 256 + threadIdx.x;
    if (i >= VOL) return;
    z[i] = t[i] + lam[c] * r[i % VV] * (1.0f / (float)V);
}

// circulant forward differences pn=dx, qn=dy, sn=dz
__global__ void k_grad(const float* __restrict__ z, float* pn, float* qn, float* sn) {
    int i = blockIdx.x * 256 + threadIdx.x;
    if (i >= VOL) return;
    int xx = i % V, yy = (i / V) % V, zz = i / VV;
    int xp = (xx + 1 == V) ? 0 : xx + 1;
    int yp = (yy + 1 == V) ? 0 : yy + 1;
    int zp = (zz + 1 == V) ? 0 : zz + 1;
    float zv = z[i];
    pn[i] = z[(zz * V + yy) * V + xp] - zv;
    qn[i] = z[(zz * V + yp) * V + xx] - zv;
    sn[i] = z[(zp * V + yy) * V + xx] - zv;
}

// four 1->8 convs + relu, concatenated -> feat [z][y][x][32] in bf16
__global__ void k_edgeconv(const float* __restrict__ pn, const float* __restrict__ qn,
                           const float* __restrict__ sn, const float* __restrict__ z,
                           const float* __restrict__ w1, const float* __restrict__ b1,
                           const float* __restrict__ w2, const float* __restrict__ b2,
                           const float* __restrict__ w3, const float* __restrict__ b3,
                           const float* __restrict__ w4, const float* __restrict__ b4,
                           u16* __restrict__ feat) {
    int i = blockIdx.x * 256 + threadIdx.x;
    if (i >= VOL) return;
    int xx = i % V, yy = (i / V) % V, zz = i / VV;
    const float* srcs[4] = { pn, qn, sn, z };
    const float* ws[4]   = { w1, w2, w3, w4 };
    const float* bs[4]   = { b1, b2, b3, b4 };
    u16* dst = feat + (size_t)i * C4;
    float nb[27];
    for (int g = 0; g < 4; ++g) {
        const float* src = srcs[g];
        int k = 0;
        for (int kd = 0; kd < 3; ++kd)
            for (int kh = 0; kh < 3; ++kh)
                for (int kw = 0; kw < 3; ++kw, ++k) {
                    int zi = zz + kd - 1, yi = yy + kh - 1, xi = xx + kw - 1;
                    bool in = ((unsigned)zi < V) && ((unsigned)yi < V) && ((unsigned)xi < V);
                    nb[k] = in ? src[(zi * V + yi) * V + xi] : 0.f;  // SAME zero-pad
                }
        for (int oc = 0; oc < 8; ++oc) {
            float a = bs[g][oc];
            const float* w = ws[g] + oc * 27;
            for (int k2 = 0; k2 < 27; ++k2) a = fmaf(nb[k2], w[k2], a);
            a = a > 0.f ? a : 0.f;
            dst[g * 8 + oc] = f2bf(a);
        }
    }
}

// wb [o][i][27] f32  ->  wt [tap][o][i] bf16  (B-fragment friendly: K=in contiguous)
__global__ void k_prepw(const float* __restrict__ wb, u16* __restrict__ wt) {
    int idx = blockIdx.x * 256 + threadIdx.x;
    if (idx >= 32 * 32 * 27) return;
    int tap = idx % 27;
    int ic  = (idx / 27) % 32;
    int oc  = idx / (27 * 32);
    wt[((size_t)tap * 32 + oc) * 32 + ic] = f2bf(wb[idx]);
}

// ---------------------------------------------------------------------------
// 32->32 3x3x3 conv as implicit GEMM with v_wmma_f32_16x16x32_bf16.
// Block = 256 threads (8 waves), tile = 16x16 (y,x) at fixed z.
// Wave w computes rows {2w, 2w+1}; each row: M=16 x-voxels, N=32 outs.
// 27 taps * 2 micro * 2 ntiles = 108 WMMA per wave.
// Input tile staged via GLOBAL_LOAD_ASYNC_TO_LDS_B128 when available; output
// repacked through LDS and streamed with coalesced / async b128 stores.
// ---------------------------------------------------------------------------
__global__ __launch_bounds__(256, 1)
void k_conv32(const u16* __restrict__ in, const u16* __restrict__ wt,
              const float* __restrict__ bias, u16* __restrict__ out, int relu) {
    __shared__ union {
        u16 in_t[3 * TIN * TIN * C4];   // 62208 B, [lz][ly][lx][ch] bf16
        u16 out_t[16 * 16 * C4];        // 16384 B, [ly][lx][ch] bf16
    } lds;

    const int tid = threadIdx.x;
    const int gx0 = blockIdx.x * 16;
    const int gy0 = blockIdx.y * 16;
    const int gz  = blockIdx.z;

    // warm the L2 path for the weight table (reused by every block)
    __builtin_prefetch(wt + (size_t)tid * 64, 0, 3);

    // stage input tile (+halo, zero-padded) into LDS, 64B per voxel
    for (int v = tid; v < 3 * TIN * TIN; v += 256) {
        int lx = v % TIN, ly = (v / TIN) % TIN, lz = v / (TIN * TIN);
        int gxx = gx0 + lx - 1, gyy = gy0 + ly - 1, gzz = gz + lz - 1;
        u16* d = &lds.in_t[(size_t)v * C4];
        if ((unsigned)gxx < V && (unsigned)gyy < V && (unsigned)gzz < V) {
            const u16* src = in + ((size_t)(gzz * V + gyy) * V + gxx) * C4;
#if defined(USE_ASYNC_LD)
            g_v4i_p gsrc = (g_v4i_p)src;
            l_v4i_p ldst = (l_v4i_p)d;
            __builtin_amdgcn_global_load_async_to_lds_b128(gsrc + 0, ldst + 0, 0, 0);
            __builtin_amdgcn_global_load_async_to_lds_b128(gsrc + 1, ldst + 1, 0, 0);
            __builtin_amdgcn_global_load_async_to_lds_b128(gsrc + 2, ldst + 2, 0, 0);
            __builtin_amdgcn_global_load_async_to_lds_b128(gsrc + 3, ldst + 3, 0, 0);
#else
            const uint4* s4 = (const uint4*)src;
            uint4* d4 = (uint4*)d;
            d4[0] = s4[0]; d4[1] = s4[1]; d4[2] = s4[2]; d4[3] = s4[3];
#endif
        } else {
            uint4 zed = {0u, 0u, 0u, 0u};
            uint4* d4 = (uint4*)d;
            d4[0] = zed; d4[1] = zed; d4[2] = zed; d4[3] = zed;
        }
    }
#if defined(USE_ASYNC_LD)
    __builtin_amdgcn_s_wait_asynccnt(0);
#endif
    __syncthreads();

    const int lane = tid & 31;
    const int wv   = tid >> 5;    // wave id 0..7
    const int lm   = lane & 15;   // M (A) / N (B,D) column
    const int kh   = lane >> 4;   // K-half selector per ISA 16-bit layouts

    union Frag { uint4 u[2]; bf16x16 v; };

    f32x8 acc[2][2];
    for (int n = 0; n < 2; ++n) {
        float bv = bias[n * 16 + lm];
        f32x8 a;
        for (int e = 0; e < 8; ++e) a[e] = bv;
        acc[0][n] = a; acc[1][n] = a;
    }

    for (int tap = 0; tap < 27; ++tap) {
        const int kw  = tap % 3;
        const int khh = (tap / 3) % 3;
        const int kd  = tap / 9;

        // B fragments (32x16 each): [tap][N][K] in global (L2-resident), K contiguous
        Frag b0, b1;
        {
            const u16*   wb = wt + (size_t)tap * 32 * 32;
            const uint4* p0 = (const uint4*)(wb + ((size_t)lm * 32 + kh * 8));
            b0.u[0] = p0[0]; b0.u[1] = p0[2];          // K and K+16 halves
            const uint4* p1 = (const uint4*)(wb + (((size_t)16 + lm) * 32 + kh * 8));
            b1.u[0] = p1[0]; b1.u[1] = p1[2];
        }

        for (int m = 0; m < 2; ++m) {
            const int ly = 2 * wv + m + khh;   // y+dy (+1 halo offset folded in)
            const int lx = lm + kw;            // x+dx
            const int lz = kd;                 // z+dz
            Frag a;
            const u16*   ab = &lds.in_t[(((size_t)lz * TIN + ly) * TIN + lx) * C4 + kh * 8];
            const uint4* pa = (const uint4*)ab;
            a.u[0] = pa[0]; a.u[1] = pa[2];    // K and K+16 halves (ds_load_b128 x2)

            acc[m][0] = __builtin_amdgcn_wmma_f32_16x16x32_bf16(
                false, a.v, false, b0.v, (short)0, acc[m][0], false, false);
            acc[m][1] = __builtin_amdgcn_wmma_f32_16x16x32_bf16(
                false, a.v, false, b1.v, (short)0, acc[m][1], false, false);
        }
    }

    // repack D fragments (lane: M = r + 8*kh, N = lm) through LDS, then emit
    // fully-coalesced 64B/voxel stores.
    __syncthreads();   // everyone done reading lds.in_t (aliased below)
    for (int m = 0; m < 2; ++m) {
        const int row = 2 * wv + m;
        for (int n = 0; n < 2; ++n) {
            const int ch = n * 16 + lm;
            for (int r = 0; r < 8; ++r) {
                float vv = acc[m][n][r];
                if (relu) vv = vv > 0.f ? vv : 0.f;
                lds.out_t[((size_t)row * 16 + (r + 8 * kh)) * C4 + ch] = f2bf(vv);
            }
        }
    }
    __syncthreads();
    {
        const int ly = tid >> 4, lx = tid & 15;  // one voxel per thread
        u16* dst = out + ((size_t)(gz * V + (gy0 + ly)) * V + (gx0 + lx)) * C4;
        u16* src = &lds.out_t[(size_t)tid * C4];
#if defined(USE_ASYNC_ST)
        g_v4i_p gdst = (g_v4i_p)dst;
        l_v4i_p lsrc = (l_v4i_p)src;
        __builtin_amdgcn_global_store_async_from_lds_b128(gdst + 0, lsrc + 0, 0, 0);
        __builtin_amdgcn_global_store_async_from_lds_b128(gdst + 1, lsrc + 1, 0, 0);
        __builtin_amdgcn_global_store_async_from_lds_b128(gdst + 2, lsrc + 2, 0, 0);
        __builtin_amdgcn_global_store_async_from_lds_b128(gdst + 3, lsrc + 3, 0, 0);
        __builtin_amdgcn_s_wait_asynccnt(0);
#else
        const uint4* s4 = (const uint4*)src;
        uint4* d4 = (uint4*)dst;
        d4[0] = s4[0]; d4[1] = s4[1]; d4[2] = s4[2]; d4[3] = s4[3];
#endif
    }
}

// ---------------------------------------------------------------------------
// HAB gates (four 8->1 convs + sigmoid) and state updates; zt = t + nt*(t-znew)
// Each gate's 8 input channels are one uint4 (16B of bf16) -> b128 loads.
__global__ void k_hab(const u16* __restrict__ outb, const float* __restrict__ z,
                      const float* __restrict__ t,
                      const float* __restrict__ pn, const float* __restrict__ qn,
                      const float* __restrict__ sn,
                      float* __restrict__ p, float* __restrict__ q, float* __restrict__ s,
                      float* __restrict__ zt,
                      const float* __restrict__ wp, const float* __restrict__ bp,
                      const float* __restrict__ wq, const float* __restrict__ bq,
                      const float* __restrict__ wsg, const float* __restrict__ bsg,
                      const float* __restrict__ wz, const float* __restrict__ bz,
                      const float* __restrict__ ntx, const float* __restrict__ nty,
                      const float* __restrict__ ntz, const float* __restrict__ nt, int c) {
    int i = blockIdx.x * 256 + threadIdx.x;
    if (i >= VOL) return;
    int xx = i % V, yy = (i / V) % V, zz = i / VV;

    float g[4] = { bp[0], bq[0], bsg[0], bz[0] };
    const float* wg[4] = { wp, wq, wsg, wz };

    int k = 0;
    for (int kd = 0; kd < 3; ++kd)
        for (int kh2 = 0; kh2 < 3; ++kh2)
            for (int kw = 0; kw < 3; ++kw, ++k) {
                int zi = zz + kd - 1, yi = yy + kh2 - 1, xi = xx + kw - 1;
                if ((unsigned)zi >= V || (unsigned)yi >= V || (unsigned)xi >= V) continue;
                const uint4* ob4 = (const uint4*)(outb + ((size_t)(zi * V + yi) * V + xi) * C4);
                for (int gi = 0; gi < 4; ++gi) {
                    uint4 qv = ob4[gi];
                    const float* wgi = wg[gi];
                    unsigned cw[4] = { qv.x, qv.y, qv.z, qv.w };
                    float a = 0.f;
                    for (int j = 0; j < 4; ++j) {
                        a = fmaf(bf2f((u16)(cw[j] & 0xFFFFu)), wgi[(2 * j + 0) * 27 + k], a);
                        a = fmaf(bf2f((u16)(cw[j] >> 16)),     wgi[(2 * j + 1) * 27 + k], a);
                    }
                    g[gi] += a;
                }
            }
    for (int gi = 0; gi < 4; ++gi) g[gi] = 1.f / (1.f + __expf(-g[gi]));

    float zv = z[i], tv = t[i];
    float p_ = pn[i] * g[0];
    float q_ = qn[i] * g[1];
    float s_ = sn[i] * g[2];
    float znew = zv * g[3] + zv;
    float pv = p[i], qv2 = q[i], sv = s[i];
    p[i] = pv + ntx[c] * (pv - p_);
    q[i] = qv2 + nty[c] * (qv2 - q_);
    s[i] = sv + ntz[c] * (sv - s_);
    zt[i] = tv + nt[c] * (tv - znew);
}

// t = gradT(q,y) + gradT(p,x) + gradT(s,z) + zt   (gradT: roll +1 minus self)
__global__ void k_assemble(const float* __restrict__ p, const float* __restrict__ q,
                           const float* __restrict__ s, const float* __restrict__ zt,
                           float* __restrict__ t) {
    int i = blockIdx.x * 256 + threadIdx.x;
    if (i >= VOL) return;
    int xx = i % V, yy = (i / V) % V, zz = i / VV;
    int xm = (xx == 0) ? V - 1 : xx - 1;
    int ym = (yy == 0) ? V - 1 : yy - 1;
    int zm = (zz == 0) ? V - 1 : zz - 1;
    t[i] = (q[(zz * V + ym) * V + xx] - q[i]) +
           (p[(zz * V + yy) * V + xm] - p[i]) +
           (s[(zm * V + yy) * V + xx] - s[i]) + zt[i];
}

__global__ void k_copyout(const float* __restrict__ t, float* __restrict__ o) {
    int i = blockIdx.x * 256 + threadIdx.x;
    if (i < VOL) o[i] = t[i];
}

// ---------------------------------------------------------------------------
extern "C" void kernel_launch(void* const* d_in, const int* in_sizes, int n_in,
                              void* d_out, int out_size, void* d_ws, size_t ws_size,
                              hipStream_t stream) {
    (void)in_sizes; (void)n_in; (void)out_size; (void)ws_size;

    const float* image = (const float*)d_in[0];
    const float* sino  = (const float*)d_in[1];
    const float* w1 = (const float*)d_in[2];  const float* b1 = (const float*)d_in[3];
    const float* w2 = (const float*)d_in[4];  const float* b2 = (const float*)d_in[5];
    const float* w3 = (const float*)d_in[6];  const float* b3 = (const float*)d_in[7];
    const float* w4 = (const float*)d_in[8];  const float* b4 = (const float*)d_in[9];
    const float* wb1 = (const float*)d_in[10]; const float* bb1 = (const float*)d_in[11];
    const float* wb2 = (const float*)d_in[12]; const float* bb2 = (const float*)d_in[13];
    const float* wp = (const float*)d_in[14]; const float* bp = (const float*)d_in[15];
    const float* wq = (const float*)d_in[16]; const float* bq = (const float*)d_in[17];
    const float* wsg = (const float*)d_in[18]; const float* bsg = (const float*)d_in[19];
    const float* wz = (const float*)d_in[20]; const float* bz = (const float*)d_in[21];
    const float* ntx = (const float*)d_in[22]; const float* nty = (const float*)d_in[23];
    const float* ntz = (const float*)d_in[24]; const float* nt  = (const float*)d_in[25];
    const float* lam = (const float*)d_in[26];
    float* outp = (float*)d_out;

    // workspace carve-up (256B-aligned)
    char* w = (char*)d_ws;
    size_t off = 0;
    auto carve = [&](size_t bytes) -> char* {
        char* r = w + off;
        off = (off + bytes + 255) & ~(size_t)255;
        return r;
    };
    float* t  = (float*)carve(sizeof(float) * VOL);
    float* z  = (float*)carve(sizeof(float) * VOL);
    float* zt = (float*)carve(sizeof(float) * VOL);
    float* p  = (float*)carve(sizeof(float) * VOL);
    float* q  = (float*)carve(sizeof(float) * VOL);
    float* s  = (float*)carve(sizeof(float) * VOL);
    float* pn = (float*)carve(sizeof(float) * VOL);
    float* qn = (float*)carve(sizeof(float) * VOL);
    float* sn = (float*)carve(sizeof(float) * VOL);
    float* r  = (float*)carve(sizeof(float) * VV);
    u16* feat = (u16*)carve(sizeof(u16) * (size_t)VOL * C4);
    u16* hbuf = (u16*)carve(sizeof(u16) * (size_t)VOL * C4);
    u16* outb = (u16*)carve(sizeof(u16) * (size_t)VOL * C4);
    u16* wt1  = (u16*)carve(sizeof(u16) * 27 * 32 * 32);
    u16* wt2  = (u16*)carve(sizeof(u16) * 27 * 32 * 32);

    const int BV = (VOL + 255) / 256;       // 3456
    const dim3 gconv(V / 16, V / 16, V);    // 6 x 6 x 96

    k_init<<<BV, 256, 0, stream>>>(image, t, p, q, s);
    k_prepw<<<(27 * 32 * 32 + 255) / 256, 256, 0, stream>>>(wb1, wt1);
    k_prepw<<<(27 * 32 * 32 + 255) / 256, 256, 0, stream>>>(wb2, wt2);

    for (int c = 0; c < 3; ++c) {
        k_residual<<<(VV + 255) / 256, 256, 0, stream>>>(t, sino, r);
        k_z<<<BV, 256, 0, stream>>>(t, r, lam, c, z);
        k_grad<<<BV, 256, 0, stream>>>(z, pn, qn, sn);
        k_edgeconv<<<BV, 256, 0, stream>>>(pn, qn, sn, z,
                                           w1, b1, w2, b2, w3, b3, w4, b4, feat);
        k_conv32<<<gconv, 256, 0, stream>>>(feat, wt1, bb1, hbuf, 1);
        k_conv32<<<gconv, 256, 0, stream>>>(hbuf, wt2, bb2, outb, 0);
        k_hab<<<BV, 256, 0, stream>>>(outb, z, t, pn, qn, sn, p, q, s, zt,
                                      wp, bp, wq, bq, wsg, bsg, wz, bz,
                                      ntx, nty, ntz, nt, c);
        k_assemble<<<BV, 256, 0, stream>>>(p, q, s, zt, t);
    }
    k_copyout<<<BV, 256, 0, stream>>>(t, outp);
}
